// SpectralRefinement_28621662060963
// MI455X (gfx1250) — compile-verified, matrix operand order checked by reading, not run on hardware
//
#include <hip/hip_runtime.h>
#include <math.h>

// ---------------------------------------------------------------------------
// SpectralRefinement for MI455X (gfx1250, wave32, WMMA)
//
// Pipeline (all f32 in/out, bf16 only as WMMA operand precision):
//   k_front : LN_in -> conv_in(2->16) -> GELU -> x ; gx = Wx @ x  (WMMA, packed)
//   k_gru   : sequential scan over T=1000; 48 single-wave WGs, h in regs,
//             3x v_wmma per step via an LDS bounce of h. gx stream is
//             software-pipelined (load t+1 while computing t) so the only
//             serial latency per step is DS-bounce + WMMA + gate math.
//   k_attn  : per (b,t): x+hs -> LN -> QKV (WMMA) -> local scores (VALU,
//             block-diag 4x4) + long scores (WMMA) -> softmax -> local O
//             (VALU) + long O (WMMA) -> out-proj + residual.
//   k_mlp   : per (b,t): LN -> W1 (WMMA) -> GELU -> W2 (WMMA) -> residual.
//   k_out   : LN -> conv_out(16->2) -> tanh -> input residual -> d_out.
// ---------------------------------------------------------------------------

#define Bb 8
#define Tt 1000
#define Ff 96
#define Cc 16
#define Dd 64

typedef __attribute__((ext_vector_type(16))) __bf16 v16bf;
typedef __attribute__((ext_vector_type(8)))  float  v8f;

#define WMMA_BF16(A_, B_, C_) \
  __builtin_amdgcn_wmma_f32_16x16x32_bf16(false, (A_), false, (B_), (short)0, (C_), false, false)

// --- Fragment loaders (CDNA5 16-bit A/B layout; CDNA5 ISA) -------
// A 16x32 (and B 32x16 mirrored): lane L holds row/col (L&15);
// half-elements e<8 -> K = 8*(L>>4)+e ; e>=8 -> K = 16 + 8*(L>>4) + (e-8).
// D/C 16x16 f32: elem i of lane L -> (M = i + 8*(L>>4), N = L&15).

__device__ __forceinline__ v16bf load_frag_f32(const float* base, int ld, int mn0, int kmax) {
  int lane = threadIdx.x & 31;
  int mn = (lane & 15) + mn0;
  int kb = (lane >> 4) * 8;
  v16bf r;
#pragma unroll
  for (int e = 0; e < 16; ++e) {
    int k = (e < 8) ? (kb + e) : (16 + kb + (e - 8));
    r[e] = (k < kmax) ? (__bf16)base[mn * ld + k] : (__bf16)0.0f;
  }
  return r;
}

__device__ __forceinline__ v16bf load_frag_bf16(const __bf16* base, int ld, int mn0, int kmax) {
  int lane = threadIdx.x & 31;
  int mn = (lane & 15) + mn0;
  int kb = (lane >> 4) * 8;
  v16bf r;
#pragma unroll
  for (int e = 0; e < 16; ++e) {
    int k = (e < 8) ? (kb + e) : (16 + kb + (e - 8));
    r[e] = (k < kmax) ? base[mn * ld + k] : (__bf16)0.0f;
  }
  return r;
}

// element (k, mn) fetched from storage base[k*ld + mn]  (k-major storage)
__device__ __forceinline__ v16bf load_frag_bf16_kmaj(const __bf16* base, int ld, int mn0, int kmax) {
  int lane = threadIdx.x & 31;
  int mn = (lane & 15) + mn0;
  int kb = (lane >> 4) * 8;
  v16bf r;
#pragma unroll
  for (int e = 0; e < 16; ++e) {
    int k = (e < 8) ? (kb + e) : (16 + kb + (e - 8));
    r[e] = (k < kmax) ? base[k * ld + mn] : (__bf16)0.0f;
  }
  return r;
}

__device__ __forceinline__ float gelu_erf(float a) {
  return 0.5f * a * (1.0f + erff(a * 0.70710678118654752f));
}

// ---------------------------------------------------------------------------
// k_front: LN_in + conv_in + GELU -> x (global f32) ; gx = Wx @ x (WMMA),
// stored bf16 in fragment-packed order [t][b][gt][ft][lane][8].
// ---------------------------------------------------------------------------
__global__ __launch_bounds__(256) void k_front(
    const float* __restrict__ in, const float* __restrict__ lng, const float* __restrict__ lnb,
    const float* __restrict__ w_in, const float* __restrict__ b_in,
    const float* __restrict__ wx, float* __restrict__ xg, __bf16* __restrict__ gx) {
  int t = blockIdx.x % Tt, b = blockIdx.x / Tt;
  __shared__ float  yin[2 * Ff];
  __shared__ float  xs[Cc * Ff];
  __shared__ __bf16 xb[Cc * Ff];
  int tid = threadIdx.x, lane = tid & 31, wave = tid >> 5;

  for (int e = tid; e < 2 * Ff; e += 256) {
    int c = e / Ff, f = e % Ff;
    yin[e] = in[(((size_t)b * 2 + c) * Tt + t) * Ff + f];
  }
  __syncthreads();
  if (tid < 2) {
    float m = 0.f;
    for (int f = 0; f < Ff; ++f) m += yin[tid * Ff + f];
    m *= (1.0f / Ff);
    float v = 0.f;
    for (int f = 0; f < Ff; ++f) { float d = yin[tid * Ff + f] - m; v += d * d; }
    float inv = rsqrtf(v * (1.0f / Ff) + 1e-5f);
    for (int f = 0; f < Ff; ++f)
      yin[tid * Ff + f] = (yin[tid * Ff + f] - m) * inv * lng[f] + lnb[f];
  }
  __syncthreads();
  for (int e = tid; e < Cc * Ff; e += 256) {
    int c = e / Ff, f = e % Ff;
    float a = w_in[c * 2 + 0] * yin[f] + w_in[c * 2 + 1] * yin[Ff + f] + b_in[c];
    float g = gelu_erf(a);
    xs[e] = g;
    xb[e] = (__bf16)g;
    xg[(((size_t)b * Cc + c) * Tt + t) * Ff + f] = g;
  }
  __syncthreads();
  // gx^T = x^T(96x16) @ Wx^T(16x48): 6 f-tiles x 3 gate-tiles of WMMA
  for (int j = wave; j < 18; j += 8) {
    int ft = j / 3, gt = j % 3;
    v16bf A = load_frag_bf16_kmaj(xb, Ff, ft * 16, Cc);  // (m=f,k=c) from xb[c][f]
    v16bf Bm = load_frag_f32(wx, Cc, gt * 16, Cc);       // (k=c,n=g) from wx[g][c]
    v8f acc = {};
    acc = WMMA_BF16(A, Bm, acc);
    size_t base = (((((size_t)t * Bb + b) * 3 + gt) * 6 + ft) * 32 + lane) * 8;
#pragma unroll
    for (int i = 0; i < 8; ++i) gx[base + i] = (__bf16)acc[i];
  }
}

// ---------------------------------------------------------------------------
// k_gru: 48 single-wave workgroups (b x f-tile). Sequential scan; h held in
// D-layout registers, bounced through LDS each step to form the A fragment.
// gx stream double-buffered: step t+1 loads issue before step t's WMMA/gate
// math, so their s_wait lands after the compute instead of before it.
// ---------------------------------------------------------------------------
__global__ __launch_bounds__(32) void k_gru(
    const float* __restrict__ wh, const __bf16* __restrict__ gx,
    float* __restrict__ hs, float* __restrict__ hlast) {
  int b = blockIdx.x / 6, ft = blockIdx.x % 6;
  int lane = threadIdx.x;
  __shared__ float hl[Cc * Cc];  // [f_local][c]

  v16bf B0 = load_frag_f32(wh, Cc, 0,  Cc);   // (k=c,n=g) gates 0..15  (z)
  v16bf B1 = load_frag_f32(wh, Cc, 16, Cc);   // gates 16..31 (r)
  v16bf B2 = load_frag_f32(wh, Cc, 32, Cc);   // gates 32..47 (n)

  float h[8];
#pragma unroll
  for (int i = 0; i < 8; ++i) h[i] = 0.f;

  // pre-load gx for t = 0
  float cur[3][8], nxt[3][8];
#pragma unroll
  for (int gt = 0; gt < 3; ++gt) {
    size_t base = ((((size_t)0 * Bb + b) * 3 + gt) * 6 + ft) * 32 * 8 + (size_t)lane * 8;
#pragma unroll
    for (int i = 0; i < 8; ++i) cur[gt][i] = (float)gx[base + i];
  }

  for (int t = 0; t < Tt; ++t) {
#pragma unroll
    for (int i = 0; i < 8; ++i)
      hl[(i + 8 * (lane >> 4)) * Cc + (lane & 15)] = h[i];
    __syncthreads();
    v16bf A = load_frag_f32(hl, Cc, 0, Cc);   // H^T: (m=f_local, k=c)
    __syncthreads();

    // issue next step's gate-input loads now; consumed after rotate below
    if (t + 1 < Tt) {
#pragma unroll
      for (int gt = 0; gt < 3; ++gt) {
        size_t base = (((((size_t)(t + 1) * Bb + b) * 3 + gt) * 6 + ft) * 32 + lane) * 8;
#pragma unroll
        for (int i = 0; i < 8; ++i) nxt[gt][i] = (float)gx[base + i];
      }
      if (t + 2 < Tt) {  // keep the stream ahead in L0/L2
        size_t pb = (((((size_t)(t + 2) * Bb + b) * 3) * 6 + ft) * 32 + lane) * 8;
        __builtin_prefetch(&gx[pb], 0, 3);
      }
    }

    v8f d0 = {}, d1 = {}, d2 = {};
    d0 = WMMA_BF16(A, B0, d0);
    d1 = WMMA_BF16(A, B1, d1);
    d2 = WMMA_BF16(A, B2, d2);

#pragma unroll
    for (int i = 0; i < 8; ++i) {
      float z = 1.0f / (1.0f + __expf(-(cur[0][i] + d0[i])));
      float r = 1.0f / (1.0f + __expf(-(cur[1][i] + d1[i])));
      float n = tanhf(cur[2][i] + r * d2[i]);
      h[i] = (1.0f - z) * n + z * h[i];
    }
    size_t hb = ((((size_t)t * Bb + b) * 6 + ft) * 32 + lane) * 8;
#pragma unroll
    for (int i = 0; i < 8; ++i) hs[hb + i] = h[i];

    // rotate double buffer (wait for t+1 loads happens here, after compute)
    if (t + 1 < Tt) {
#pragma unroll
      for (int gt = 0; gt < 3; ++gt)
#pragma unroll
        for (int i = 0; i < 8; ++i) cur[gt][i] = nxt[gt][i];
    }
  }
  // h_last -> d_out tail: [B][C][F]
#pragma unroll
  for (int i = 0; i < 8; ++i) {
    int c = lane & 15, f = ft * 16 + i + 8 * (lane >> 4);
    hlast[((size_t)b * Cc + c) * Ff + f] = h[i];
  }
}

// ---------------------------------------------------------------------------
// k_attn: per (b,t) tile. Dynamic LDS (aliased, 65024 B).
// ---------------------------------------------------------------------------
__global__ __launch_bounds__(256) void k_attn(
    float* __restrict__ xg, const float* __restrict__ hs,
    const float* __restrict__ lng, const float* __restrict__ lnb,
    const float* __restrict__ wq, const float* __restrict__ bq,
    const float* __restrict__ wk, const float* __restrict__ bk,
    const float* __restrict__ wv, const float* __restrict__ bv,
    const float* __restrict__ wo, const float* __restrict__ bo) {
  int t = blockIdx.x % Tt, b = blockIdx.x / Tt;
  extern __shared__ char smem[];
  float*  xs  = (float*)(smem);              // [16][96]  f32   6144
  __bf16* ybf = (__bf16*)(smem + 6144);      // [16][96]  bf16  3072
  __bf16* Qb  = (__bf16*)(smem + 9216);      // [96][64]  bf16 12288
  __bf16* Kb  = (__bf16*)(smem + 21504);     // [96][64]  bf16 12288
  __bf16* Vb  = (__bf16*)(smem + 33792);     // [96][64]  bf16 12288
  __bf16* KL  = (__bf16*)(smem + 46080);     // [32][64]  bf16  4096 (rows>=24 zero)
  __bf16* VL  = (__bf16*)(smem + 50176);     // [32][64]  bf16  4096
  float*  SC  = (float*)(smem + 54272);      // [96][28]  f32  10752
  float*  OS  = (float*)(smem + 9216);       // [96][64]  f32  24576 (aliases Qb+Kb)
  int tid = threadIdx.x, lane = tid & 31, wave = tid >> 5;
  const float scale = 0.125f;  // D^-0.5

  // x_tile = x + hs (GRU residual); hs is fragment-packed
  for (int e = tid; e < Cc * Ff; e += 256) {
    int c = e / Ff, f = e % Ff;
    int ftl = f >> 4, fl = f & 15, hlane = c + 16 * (fl >> 3), hi = fl & 7;
    float hv = hs[((((size_t)t * Bb + b) * 6 + ftl) * 32 + hlane) * 8 + hi];
    xs[e] = xg[(((size_t)b * Cc + c) * Tt + t) * Ff + f] + hv;
  }
  __syncthreads();
  if (tid < Cc) {  // LayerNorm over f per channel
    float m = 0.f;
    for (int f = 0; f < Ff; ++f) m += xs[tid * Ff + f];
    m *= (1.0f / Ff);
    float v = 0.f;
    for (int f = 0; f < Ff; ++f) { float d = xs[tid * Ff + f] - m; v += d * d; }
    float inv = rsqrtf(v * (1.0f / Ff) + 1e-5f);
    for (int f = 0; f < Ff; ++f)
      ybf[tid * Ff + f] = (__bf16)((xs[tid * Ff + f] - m) * inv * lng[f] + lnb[f]);
  }
  __syncthreads();
  // Q/K/V = y^T(96x16) @ W^T(16x64) + bias : 3 x 6 x 4 WMMA tiles
  for (int j = wave; j < 72; j += 8) {
    int mat = j / 24, r = j % 24, ft = r / 4, dt = r % 4;
    const float* W  = (mat == 0) ? wq : (mat == 1) ? wk : wv;
    const float* Bv = (mat == 0) ? bq : (mat == 1) ? bk : bv;
    __bf16* Dst     = (mat == 0) ? Qb : (mat == 1) ? Kb : Vb;
    v16bf A  = load_frag_bf16_kmaj(ybf, Ff, ft * 16, Cc);
    v16bf Bf = load_frag_f32(W, Cc, dt * 16, Cc);
    v8f acc = {};
    acc = WMMA_BF16(A, Bf, acc);
    int d_ = dt * 16 + (lane & 15);
    float bias = Bv[d_];
#pragma unroll
    for (int i = 0; i < 8; ++i) {
      int f_ = ft * 16 + i + 8 * (lane >> 4);
      Dst[f_ * Dd + d_] = (__bf16)(acc[i] + bias);
    }
  }
  __syncthreads();
  // segment-pooled long K/V (mean over R=4), zero-padded to 32 rows
  for (int e = tid; e < 32 * Dd; e += 256) {
    int m = e >> 6, d = e & 63;
    float kv = 0.f, vv = 0.f;
    if (m < 24) {
#pragma unroll
      for (int kk = 0; kk < 4; ++kk) {
        kv += (float)Kb[(4 * m + kk) * Dd + d];
        vv += (float)Vb[(4 * m + kk) * Dd + d];
      }
      kv *= 0.25f; vv *= 0.25f;
    }
    KL[e] = (__bf16)kv; VL[e] = (__bf16)vv;
  }
  __syncthreads();
  // local scores (block-diagonal 4x4 per segment) — VALU
  for (int j = tid; j < 384; j += 256) {
    int q = j >> 2, kk = j & 3, key = (q >> 2) * 4 + kk;
    float s = 0.f;
    for (int d = 0; d < Dd; ++d) s += (float)Qb[q * Dd + d] * (float)Kb[key * Dd + d];
    SC[q * 28 + kk] = s * scale;
  }
  // long scores: Q(96x64) @ KL^T(64x24) — WMMA, K chunked 2x32
  for (int j = wave; j < 12; j += 8) {
    int ft = j / 2, nt = j % 2;
    v8f acc = {};
#pragma unroll
    for (int kc = 0; kc < 2; ++kc) {
      v16bf A  = load_frag_bf16(Qb + kc * 32, Dd, ft * 16, 32);
      v16bf Bf = load_frag_bf16(KL + kc * 32, Dd, nt * 16, 32);
      acc = WMMA_BF16(A, Bf, acc);
    }
    int m_ = nt * 16 + (lane & 15);
    if (m_ < 24) {
#pragma unroll
      for (int i = 0; i < 8; ++i) {
        int f_ = ft * 16 + i + 8 * (lane >> 4);
        SC[f_ * 28 + 4 + m_] = acc[i] * scale;
      }
    }
  }
  __syncthreads();
  // softmax over 28 (4 local + 24 long), in place
  if (tid < Ff) {
    float* row = SC + tid * 28;
    float mx = -1e30f;
    for (int k = 0; k < 28; ++k) mx = fmaxf(mx, row[k]);
    float sm = 0.f;
    for (int k = 0; k < 28; ++k) { float e = expf(row[k] - mx); row[k] = e; sm += e; }
    float inv = 1.0f / sm;
    for (int k = 0; k < 28; ++k) row[k] *= inv;
  }
  __syncthreads();
  // local output part — VALU (OS aliases dead Qb/Kb)
  for (int e = tid; e < Ff * Dd; e += 256) {
    int q = e >> 6, d = e & 63, s4 = (q >> 2) * 4;
    float o = 0.f;
#pragma unroll
    for (int kk = 0; kk < 4; ++kk) o += SC[q * 28 + kk] * (float)Vb[(s4 + kk) * Dd + d];
    OS[e] = o;
  }
  __syncthreads();
  // long output part: A_long(96x24) @ VL(24x64) — WMMA, K padded to 32
  for (int j = wave; j < 24; j += 8) {
    int ft = j / 4, dt = j % 4;
    v16bf A  = load_frag_f32(SC + 4, 28, ft * 16, 24);        // probs, zero pad k>=24
    v16bf Bf = load_frag_bf16_kmaj(VL, Dd, dt * 16, 32);      // rows 24..31 are zero
    v8f acc = {};
    acc = WMMA_BF16(A, Bf, acc);
    int d_ = dt * 16 + (lane & 15);
#pragma unroll
    for (int i = 0; i < 8; ++i) {
      int f_ = ft * 16 + i + 8 * (lane >> 4);
      OS[f_ * Dd + d_] += acc[i];
    }
  }
  __syncthreads();
  // output projection + residual -> global x
  for (int e = tid; e < Cc * Ff; e += 256) {
    int c = e / Ff, f = e % Ff;
    float a = xs[e] + bo[c];
    for (int d = 0; d < Dd; ++d) a += wo[c * Dd + d] * OS[f * Dd + d];
    xg[(((size_t)b * Cc + c) * Tt + t) * Ff + f] = a;
  }
}

// ---------------------------------------------------------------------------
// k_mlp: PreNorm conv-GELU-conv (16->16->16), WMMA both matmuls.
// ---------------------------------------------------------------------------
__global__ __launch_bounds__(256) void k_mlp(
    float* __restrict__ xg, const float* __restrict__ lng, const float* __restrict__ lnb,
    const float* __restrict__ w1, const float* __restrict__ b1,
    const float* __restrict__ w2, const float* __restrict__ b2) {
  int t = blockIdx.x % Tt, b = blockIdx.x / Tt;
  __shared__ float  xs[Cc * Ff];
  __shared__ __bf16 ybf[Cc * Ff];
  __shared__ __bf16 y1[Ff * Cc];  // [f][c']
  int tid = threadIdx.x, lane = tid & 31, wave = tid >> 5;

  for (int e = tid; e < Cc * Ff; e += 256) {
    int c = e / Ff, f = e % Ff;
    xs[e] = xg[(((size_t)b * Cc + c) * Tt + t) * Ff + f];
  }
  __syncthreads();
  if (tid < Cc) {
    float m = 0.f;
    for (int f = 0; f < Ff; ++f) m += xs[tid * Ff + f];
    m *= (1.0f / Ff);
    float v = 0.f;
    for (int f = 0; f < Ff; ++f) { float d = xs[tid * Ff + f] - m; v += d * d; }
    float inv = rsqrtf(v * (1.0f / Ff) + 1e-5f);
    for (int f = 0; f < Ff; ++f)
      ybf[tid * Ff + f] = (__bf16)((xs[tid * Ff + f] - m) * inv * lng[f] + lnb[f]);
  }
  __syncthreads();
  for (int j = wave; j < 6; j += 8) {  // y1 = GELU(y^T @ W1^T + b1)
    v16bf A  = load_frag_bf16_kmaj(ybf, Ff, j * 16, Cc);
    v16bf Bf = load_frag_f32(w1, Cc, 0, Cc);
    v8f acc = {};
    acc = WMMA_BF16(A, Bf, acc);
    int o_ = lane & 15;
    float bias = b1[o_];
#pragma unroll
    for (int i = 0; i < 8; ++i) {
      int f_ = j * 16 + i + 8 * (lane >> 4);
      y1[f_ * Cc + o_] = (__bf16)gelu_erf(acc[i] + bias);
    }
  }
  __syncthreads();
  for (int j = wave; j < 6; j += 8) {  // x += y1 @ W2^T + b2
    v16bf A  = load_frag_bf16(y1, Cc, j * 16, Cc);
    v16bf Bf = load_frag_f32(w2, Cc, 0, Cc);
    v8f acc = {};
    acc = WMMA_BF16(A, Bf, acc);
    int o_ = lane & 15;
    float bias = b2[o_];
#pragma unroll
    for (int i = 0; i < 8; ++i) {
      int f_ = j * 16 + i + 8 * (lane >> 4);
      xg[(((size_t)b * Cc + o_) * Tt + t) * Ff + f_] = xs[o_ * Ff + f_] + acc[i] + bias;
    }
  }
}

// ---------------------------------------------------------------------------
// k_out: LN -> conv_out (16->2) -> tanh -> outer residual on input.
// ---------------------------------------------------------------------------
__global__ __launch_bounds__(256) void k_out(
    const float* __restrict__ xg, const float* __restrict__ in,
    const float* __restrict__ lng, const float* __restrict__ lnb,
    const float* __restrict__ w_out, const float* __restrict__ b_out,
    float* __restrict__ out) {
  int t = blockIdx.x % Tt, b = blockIdx.x / Tt;
  __shared__ float xs[Cc * Ff];
  int tid = threadIdx.x;
  for (int e = tid; e < Cc * Ff; e += 256) {
    int c = e / Ff, f = e % Ff;
    xs[e] = xg[(((size_t)b * Cc + c) * Tt + t) * Ff + f];
  }
  __syncthreads();
  if (tid < Cc) {
    float m = 0.f;
    for (int f = 0; f < Ff; ++f) m += xs[tid * Ff + f];
    m *= (1.0f / Ff);
    float v = 0.f;
    for (int f = 0; f < Ff; ++f) { float d = xs[tid * Ff + f] - m; v += d * d; }
    float inv = rsqrtf(v * (1.0f / Ff) + 1e-5f);
    for (int f = 0; f < Ff; ++f)
      xs[tid * Ff + f] = (xs[tid * Ff + f] - m) * inv * lng[f] + lnb[f];
  }
  __syncthreads();
  for (int e = tid; e < 2 * Ff; e += 256) {
    int o = e / Ff, f = e % Ff;
    float a = b_out[o];
    for (int c = 0; c < Cc; ++c) a += w_out[o * Cc + c] * xs[c * Ff + f];
    size_t idx = (((size_t)b * 2 + o) * Tt + t) * Ff + f;
    out[idx] = in[idx] + tanhf(a);
  }
}

// ---------------------------------------------------------------------------
extern "C" void kernel_launch(void* const* d_in, const int* in_sizes, int n_in,
                              void* d_out, int out_size, void* d_ws, size_t ws_size,
                              hipStream_t stream) {
  (void)in_sizes; (void)n_in; (void)out_size; (void)ws_size;
  const float* input    = (const float*)d_in[0];
  const float* ln_in_g  = (const float*)d_in[1];
  const float* ln_in_b  = (const float*)d_in[2];
  const float* w_in     = (const float*)d_in[3];
  const float* b_in     = (const float*)d_in[4];
  const float* gru_wx   = (const float*)d_in[5];
  const float* gru_wh   = (const float*)d_in[6];
  const float* ln_att_g = (const float*)d_in[7];
  const float* ln_att_b = (const float*)d_in[8];
  const float* wq       = (const float*)d_in[9];
  const float* bq       = (const float*)d_in[10];
  const float* wk       = (const float*)d_in[11];
  const float* bk       = (const float*)d_in[12];
  const float* wv       = (const float*)d_in[13];
  const float* bv       = (const float*)d_in[14];
  const float* wo       = (const float*)d_in[15];
  const float* bo       = (const float*)d_in[16];
  const float* ln_m_g   = (const float*)d_in[17];
  const float* ln_m_b   = (const float*)d_in[18];
  const float* w_m1     = (const float*)d_in[19];
  const float* b_m1     = (const float*)d_in[20];
  const float* w_m2     = (const float*)d_in[21];
  const float* b_m2     = (const float*)d_in[22];
  const float* ln_out_g = (const float*)d_in[23];
  const float* ln_out_b = (const float*)d_in[24];
  const float* w_out    = (const float*)d_in[25];
  const float* b_out    = (const float*)d_in[26];

  // Workspace carve (172,032,000 B total):
  //   x   : [B][C][T][F] f32        49,152,000 B
  //   gx  : fragment-packed bf16    73,728,000 B
  //   hs  : fragment-packed f32     49,152,000 B
  float*  xg = (float*)d_ws;
  __bf16* gx = (__bf16*)((char*)d_ws + 49152000u);
  float*  hs = (float*)((char*)d_ws + 122880000u);

  float* out   = (float*)d_out;
  float* hlast = out + (size_t)Bb * 2 * Tt * Ff;

  k_front<<<dim3(Bb * Tt), dim3(256), 0, stream>>>(input, ln_in_g, ln_in_b, w_in, b_in,
                                                   gru_wx, xg, gx);
  k_gru<<<dim3(Bb * 6), dim3(32), 0, stream>>>(gru_wh, gx, hs, hlast);
  k_attn<<<dim3(Bb * Tt), dim3(256), 65024, stream>>>(xg, hs, ln_att_g, ln_att_b,
                                                      wq, bq, wk, bk, wv, bv, wo, bo);
  k_mlp<<<dim3(Bb * Tt), dim3(256), 0, stream>>>(xg, ln_m_g, ln_m_b, w_m1, b_m1, w_m2, b_m2);
  k_out<<<dim3(Bb * Tt), dim3(256), 0, stream>>>(xg, input, ln_out_g, ln_out_b, w_out, b_out, out);
}
